// VectorQuantizer_3985729650859
// MI455X (gfx1250) — compile-verified, matrix operand order checked by reading, not run on hardware
//
#include <hip/hip_runtime.h>

typedef _Float16 v16h __attribute__((ext_vector_type(16)));
typedef _Float16 h4   __attribute__((ext_vector_type(4)));
typedef float    v8f  __attribute__((ext_vector_type(8)));
typedef int      v4i  __attribute__((ext_vector_type(4)));

#define K_DIM   512
#define N_CODES 8192
#define M_ROWS  32768
#define M_TILE  128
#define N_TILE  128
#define K_TILE  64
#define LDS_STR 72   // halfs; 144B row stride -> 16 lanes hit disjoint 4-bank groups
#define N_CHUNKS   (N_CODES / N_TILE)       // 64
#define K_CHUNKS   (K_DIM / K_TILE)         // 8
#define TOT_CHUNKS (N_CHUNKS * K_CHUNKS)    // 512

#if defined(__HIP_DEVICE_COMPILE__) &&                                   \
    __has_builtin(__builtin_amdgcn_global_load_async_to_lds_b128) &&     \
    __has_builtin(__builtin_amdgcn_s_wait_asynccnt)
#define VQ_ASYNC 1
#else
#define VQ_ASYNC 0
#endif

#if VQ_ASYNC
// Builtin signature (from hipcc diagnostic): param0 = AS1 (global) int4*,
// param1 = AS3 (LDS) int4*, then imm offset, imm cpol.
// inttoptr casts sidestep C++ addrspace-conversion rules; AS3 pointer value is
// the low 32 bits of the generic LDS address.
#define VQ_GP(p) ((__attribute__((address_space(1))) v4i*)(unsigned long long)(p))
#define VQ_LP(p) ((__attribute__((address_space(3))) v4i*)(unsigned int)(unsigned long long)(p))
#endif

// ---- Kernel A: z f32 -> f16 (one float4 per thread) -------------------------
__global__ __launch_bounds__(256) void vq_cvt_f16(const float* __restrict__ src,
                                                  _Float16* __restrict__ dst) {
    size_t i = (size_t)blockIdx.x * 256 + threadIdx.x;
    float4 v = ((const float4*)src)[i];
    h4 h = { (_Float16)v.x, (_Float16)v.y, (_Float16)v.z, (_Float16)v.w };
    ((h4*)dst)[i] = h;
}

// ---- Kernel B: codebook f32 -> f16, fused squared norms (wave per row) ------
__global__ __launch_bounds__(256) void vq_cbprep(const float* __restrict__ cb,
                                                 _Float16* __restrict__ cbh,
                                                 float* __restrict__ norms) {
    const int wave = threadIdx.x >> 5;
    const int lane = threadIdx.x & 31;
    const int row  = blockIdx.x * 8 + wave;
    const float* src = cb  + (size_t)row * K_DIM;
    _Float16*    dst = cbh + (size_t)row * K_DIM;
    float s = 0.0f;
#pragma unroll
    for (int i = 0; i < 4; ++i) {
        int c = (i * 32 + lane) * 4;
        float4 v = *(const float4*)(src + c);
        s = fmaf(v.x, v.x, fmaf(v.y, v.y, fmaf(v.z, v.z, fmaf(v.w, v.w, s))));
        h4 h = { (_Float16)v.x, (_Float16)v.y, (_Float16)v.z, (_Float16)v.w };
        *(h4*)(dst + c) = h;
    }
#pragma unroll
    for (int m = 16; m >= 1; m >>= 1) s += __shfl_xor(s, m, 32);
    if (lane == 0) norms[row] = s;
}

// ---- Kernel C: fused GEMM + argmin ------------------------------------------
// score(row, j) = ||e_j||^2 - 2 * z_row . e_j   (||z||^2 dropped: row-constant)
__global__ __launch_bounds__(256) void vq_argmin(const _Float16* __restrict__ zh,
                                                 const _Float16* __restrict__ cbh,
                                                 const float* __restrict__ norms,
                                                 int* __restrict__ rowIdx) {
    __shared__ __align__(16) _Float16 Bs[2][N_TILE * LDS_STR];   // 2 x 18KB

    const int tid   = threadIdx.x;
    const int wave  = tid >> 5;
    const int lane  = tid & 31;
    const int laneN = lane & 15;
    const int hi    = lane >> 4;           // 0: lanes 0-15, 1: lanes 16-31
    const int blockRow = blockIdx.x * M_TILE;

    // Register-resident A: this wave's 16 rows, full K=512 -> 16 ksteps.
    // ISA 16-bit fragment layout: lanes 0-15 hold K {0..7,16..23}, lanes 16-31
    // hold K {8..15,24..31} of each 32-wide kstep.
    v16h afrag[16];
    {
        const _Float16* arow = zh + (size_t)(blockRow + wave * 16 + laneN) * K_DIM;
#pragma unroll
        for (int t = 0; t < 16; ++t) {
            const _Float16* p = arow + t * 32 + hi * 8;
            ((float4*)&afrag[t])[0] = *(const float4*)(p);
            ((float4*)&afrag[t])[1] = *(const float4*)(p + 16);
        }
    }

    float bestV[8];
    int   bestI[8];
#pragma unroll
    for (int r = 0; r < 8; ++r) { bestV[r] = 3.4e38f; bestI[r] = 0; }

    // Stage one 128x64 f16 codebook chunk into Bs[buf] (async if available).
    auto stageB = [&](int buf, int chunkId) {
        const int nc = (chunkId >> 3) << 7;   // column base
        const int kc = (chunkId & 7) << 6;    // K base
#pragma unroll
        for (int i = 0; i < 4; ++i) {
            int lin = i * 256 + tid;          // 1024 x 8-half pieces
            int r   = lin >> 3;               // 0..127 (column)
            int c8  = (lin & 7) << 3;         // 0..56  (K offset)
            const _Float16* g = cbh + (size_t)(nc + r) * K_DIM + (kc + c8);
            _Float16* l = &Bs[buf][r * LDS_STR + c8];
#if VQ_ASYNC
            __builtin_amdgcn_global_load_async_to_lds_b128(VQ_GP(g), VQ_LP(l), 0, 0);
#else
            *(float4*)l = *(const float4*)g;
#endif
        }
    };

    stageB(0, 0);

    const v8f zeroAcc = {};
    for (int ncIdx = 0; ncIdx < N_CHUNKS; ++ncIdx) {
        const int nc = ncIdx << 7;
        v8f acc[8];
#pragma unroll
        for (int s = 0; s < 8; ++s) acc[s] = zeroAcc;

#pragma unroll
        for (int kcIdx = 0; kcIdx < K_CHUNKS; ++kcIdx) {
            const int cur = kcIdx & 1;
#if VQ_ASYNC
            __builtin_amdgcn_s_wait_asynccnt(0);   // my stage of Bs[cur] landed
#endif
            __syncthreads();                       // everyone's stage landed
            {
                const int chunk = ncIdx * K_CHUNKS + kcIdx;
                if (chunk + 1 < TOT_CHUNKS) stageB(cur ^ 1, chunk + 1);
            }
#pragma unroll
            for (int ks = 0; ks < 2; ++ks) {
                const v16h a = afrag[kcIdx * 2 + ks];
                const int kSel = ks * 32 + hi * 8;
                const _Float16* bbase = &Bs[cur][laneN * LDS_STR + kSel];
                v16h bcur;
                ((float4*)&bcur)[0] = *(const float4*)(bbase);
                ((float4*)&bcur)[1] = *(const float4*)(bbase + 16);
#pragma unroll
                for (int s = 0; s < 8; ++s) {
                    v16h bnext;
                    if (s < 7) {   // prefetch next B fragment before the WMMA
                        const _Float16* bp = bbase + (s + 1) * 16 * LDS_STR;
                        ((float4*)&bnext)[0] = *(const float4*)(bp);
                        ((float4*)&bnext)[1] = *(const float4*)(bp + 16);
                    }
                    acc[s] = __builtin_amdgcn_wmma_f32_16x16x32_f16(
                        false, a, false, bcur, (short)0, acc[s], false, false);
                    if (s < 7) bcur = bnext;
                }
            }
        }

        // Fold this 128-column chunk into the running per-lane argmin.
#pragma unroll
        for (int s = 0; s < 8; ++s) {
            const int col = nc + s * 16 + laneN;
            const float nr = norms[col];
#pragma unroll
            for (int r = 0; r < 8; ++r) {
                float v = __builtin_fmaf(-2.0f, acc[s][r], nr);
                if (v < bestV[r]) { bestV[r] = v; bestI[r] = col; }
            }
        }
    }

    // Cross-lane argmin within each 16-lane half (rows r / r+8), tie -> smaller idx.
#pragma unroll
    for (int r = 0; r < 8; ++r) {
        float v = bestV[r];
        int   idx = bestI[r];
#pragma unroll
        for (int m = 1; m <= 8; m <<= 1) {
            float ov = __shfl_xor(v, m, 32);
            int   oi = __shfl_xor(idx, m, 32);
            if (ov < v || (ov == v && oi < idx)) { v = ov; idx = oi; }
        }
        if (laneN == 0) {
            int row = blockRow + wave * 16 + r + hi * 8;
            rowIdx[row] = idx;
        }
    }
}

// ---- Kernel D: gather codebook rows (f32), per-row squared error ------------
__global__ __launch_bounds__(128) void vq_gather(const float* __restrict__ z,
                                                 const float* __restrict__ cb,
                                                 const int* __restrict__ rowIdx,
                                                 float* __restrict__ qOut,
                                                 float* __restrict__ idxOut,
                                                 float* __restrict__ rowSum) {
    const int row  = blockIdx.x;
    const int t    = threadIdx.x;
    const int wave = t >> 5;
    const int lane = t & 31;
    const int idx  = rowIdx[row];

    float4 q  = ((const float4*)(cb + (size_t)idx * K_DIM))[t];
    float4 zz = ((const float4*)(z  + (size_t)row * K_DIM))[t];
    ((float4*)(qOut + (size_t)row * K_DIM))[t] = q;

    float dx = q.x - zz.x, dy = q.y - zz.y, dz = q.z - zz.z, dw = q.w - zz.w;
    float s = dx * dx + dy * dy + dz * dz + dw * dw;
#pragma unroll
    for (int m = 16; m >= 1; m >>= 1) s += __shfl_xor(s, m, 32);

    __shared__ float red[4];
    if (lane == 0) red[wave] = s;
    __syncthreads();
    if (t == 0) {
        rowSum[row] = red[0] + red[1] + red[2] + red[3];
        idxOut[row] = (float)idx;
    }
}

// ---- Kernel E: deterministic loss reduction ---------------------------------
__global__ __launch_bounds__(256) void vq_lossk(const float* __restrict__ rowSum,
                                                float* __restrict__ lossOut) {
    __shared__ float red[256];
    float s = 0.0f;
    for (int i = threadIdx.x; i < M_ROWS; i += 256) s += rowSum[i];
    red[threadIdx.x] = s;
    __syncthreads();
#pragma unroll
    for (int st = 128; st >= 1; st >>= 1) {
        if ((int)threadIdx.x < st) red[threadIdx.x] += red[threadIdx.x + st];
        __syncthreads();
    }
    if (threadIdx.x == 0)
        lossOut[0] = red[0] * (1.25f / (float)(M_ROWS * K_DIM));
}

extern "C" void kernel_launch(void* const* d_in, const int* in_sizes, int n_in,
                              void* d_out, int out_size, void* d_ws, size_t ws_size,
                              hipStream_t stream) {
    const float* zbuf = (const float*)d_in[0];   // (16, 2048, 512) f32
    const float* cb   = (const float*)d_in[1];   // (8192, 512) f32

    char* ws = (char*)d_ws;
    _Float16* zh     = (_Float16*)ws;                                  // 32 MB
    _Float16* cbh    = (_Float16*)(ws + (size_t)M_ROWS * K_DIM * 2);   // 8 MB
    char*     ws2    = ws + (size_t)(M_ROWS + N_CODES) * K_DIM * 2;
    float*    norms  = (float*)ws2;                                    // 32 KB
    int*      rowIdx = (int*)(ws2 + N_CODES * sizeof(float));          // 128 KB
    float*    rowSum = (float*)(ws2 + N_CODES * sizeof(float)
                                    + M_ROWS * sizeof(int));           // 128 KB

    float* qOut    = (float*)d_out;                   // 32768*512
    float* idxOut  = qOut + (size_t)M_ROWS * K_DIM;   // 32768
    float* lossOut = idxOut + M_ROWS;                 // 1

    vq_cvt_f16<<<(M_ROWS * K_DIM / 4) / 256, 256, 0, stream>>>(zbuf, zh);
    vq_cbprep <<<N_CODES / 8, 256, 0, stream>>>(cb, cbh, norms);
    vq_argmin <<<M_ROWS / M_TILE, 256, 0, stream>>>(zh, cbh, norms, rowIdx);
    vq_gather <<<M_ROWS, 128, 0, stream>>>(zbuf, cb, rowIdx, qOut, idxOut, rowSum);
    vq_lossk  <<<1, 256, 0, stream>>>(rowSum, lossOut);
}
